// GCN_alignment_47502338294130
// MI455X (gfx1250) — matrix operand order repaired and checked by reading, not compiled, not run on hardware
//
#include <hip/hip_runtime.h>

// ---------------------------------------------------------------------------
// Types for CDNA5 WMMA (wave32). v16bf = 8 VGPRs of packed bf16, v8f = 8 VGPRs.
// ---------------------------------------------------------------------------
typedef __bf16 bf16_t;
typedef __bf16 v16bf __attribute__((ext_vector_type(16)));
typedef __bf16 v4bf  __attribute__((ext_vector_type(4)));
typedef float  v8f   __attribute__((ext_vector_type(8)));

union FragBF { v16bf v; uint4 q[2]; };

// A-matrix fragment (16x32 bf16): lane holds row M=lane&15.
// lanes 0-15: K = k0..k0+7 then k0+16..k0+23 ; lanes 16-31: +8 on both runs.
static __device__ __forceinline__ v16bf load_afrag(const bf16_t* rowk, int lane) {
  const bf16_t* p = rowk + ((lane & 16) ? 8 : 0);
  FragBF f;
  f.q[0] = *(const uint4*)(p);
  f.q[1] = *(const uint4*)(p + 16);
  return f.v;
}

// B-matrix fragment (32x16 bf16) loaded from an N-major (transposed) row:
// lane holds column N=lane&15; lanes 0-15: K = k0..k0+7 then k0+16..k0+23 ; lanes 16-31: +8 on both runs.
static __device__ __forceinline__ v16bf load_bfrag(const bf16_t* rowk, int lane) {
  const bf16_t* p = rowk + ((lane & 16) ? 8 : 0);
  FragBF f;
  f.q[0] = *(const uint4*)(p);
  f.q[1] = *(const uint4*)(p + 16);
  return f.v;
}

static __device__ __forceinline__ v8f wmma_bf16(v16bf a, v16bf b, v8f c) {
  // D = A*B + C, f32 accumulate.  (neg_a, A, neg_b, B, c_mod, C, reuse_a, reuse_b)
  return __builtin_amdgcn_wmma_f32_16x16x32_bf16(false, a, false, b, (short)0, c,
                                                 false, false);
}

// Async global->LDS copies (CDNA5, tracked on ASYNCcnt). Per-lane addresses:
// vdst VGPR = LDS byte offset, vaddr = 64-bit global address.
static __device__ __forceinline__ unsigned lds_off(const void* p) {
  return (unsigned)(unsigned long long)p;  // LDS aperture: addr[31:0] = offset
}
static __device__ __forceinline__ void async_copy_b128(unsigned lds, const void* g) {
  asm volatile("global_load_async_to_lds_b128 %0, %1, off" ::"v"(lds), "v"(g)
               : "memory");
}
static __device__ __forceinline__ void wait_async0() {
  asm volatile("s_wait_asynccnt 0x0" ::: "memory");
}

#define V8F_ZERO {0.f,0.f,0.f,0.f,0.f,0.f,0.f,0.f}

// Problem constants
#define NN 8192
#define HH 512
#define BB 64
#define SS 3
#define WW 64

// ---------------------------------------------------------------------------
// kA_split: A (f32, row-major) -> A_hi/A_lo bf16 planes (row-major).
//           bf16x3 decomposition: x ~= hi + lo, done ONCE instead of in-loop.
// ---------------------------------------------------------------------------
__global__ __launch_bounds__(256) void kA_split(const float* __restrict__ A,
                                                bf16_t* __restrict__ Ahi,
                                                bf16_t* __restrict__ Alo) {
  size_t i4 = ((size_t)blockIdx.x * 256 + threadIdx.x) * 4;
  float4 f = *(const float4*)(A + i4);
  float fv[4] = {f.x, f.y, f.z, f.w};
  v4bf h, l;
#pragma unroll
  for (int e = 0; e < 4; e++) {
    bf16_t hi = (bf16_t)fv[e];
    h[e] = hi;
    l[e] = (bf16_t)(fv[e] - (float)hi);
  }
  *(v4bf*)(Ahi + i4) = h;
  *(v4bf*)(Alo + i4) = l;
}

// ---------------------------------------------------------------------------
// kE_splitT: E [K=8192][N=512] f32 -> EhiT/EloT [N][K] bf16 (transposed),
//            so GEMM B-tiles are K-contiguous = fragment-ready.
// ---------------------------------------------------------------------------
__global__ __launch_bounds__(256) void kE_splitT(const float* __restrict__ E,
                                                 bf16_t* __restrict__ EhiT,
                                                 bf16_t* __restrict__ EloT) {
  size_t o4 = ((size_t)blockIdx.x * 256 + threadIdx.x) * 4;  // output [n][k]
  int n = (int)(o4 / NN);
  int k = (int)(o4 % NN);
  v4bf h, l;
#pragma unroll
  for (int e = 0; e < 4; e++) {
    float x = E[(size_t)(k + e) * HH + n];
    bf16_t hi = (bf16_t)x;
    h[e] = hi;
    l[e] = (bf16_t)(x - (float)hi);
  }
  *(v4bf*)(EhiT + o4) = h;
  *(v4bf*)(EloT + o4) = l;
}

// ---------------------------------------------------------------------------
// k0: transpose Wh/Wt [S,H,H] (k-major) -> [S,H,H] (g-major) as bf16
// ---------------------------------------------------------------------------
__global__ __launch_bounds__(256) void k0_transposeW(const float* __restrict__ Wh,
                                                     const float* __restrict__ Wt,
                                                     bf16_t* __restrict__ WhT,
                                                     bf16_t* __restrict__ WtT) {
  size_t idx = (size_t)blockIdx.x * 256 + threadIdx.x;
  const size_t PER = (size_t)SS * HH * HH;
  const float* src;  bf16_t* dst;  size_t r;
  if (idx < PER) { src = Wh; dst = WhT; r = idx; }
  else           { src = Wt; dst = WtT; r = idx - PER; }
  int s   = (int)(r / (HH * HH));
  int rem = (int)(r % (HH * HH));
  int g = rem >> 9;       // 0..511
  int k = rem & 511;
  dst[r] = (bf16_t)src[(size_t)s * HH * HH + (size_t)k * HH + g];
}

// ---------------------------------------------------------------------------
// k1: layer1 = A @ embedding via bf16x3 (hi*hi + hi*lo + lo*hi), f32 accum.
//     128x128 tile / block, 8 waves each owning a 64x32 region (4x2 WMMA tiles).
//     All operands pre-split to fragment-ready bf16 planes, staged with
//     double-buffered global_load_async_to_lds_b128 (ASYNCcnt), so the inner
//     loop is pure ds_load_b128 + v_wmma with no conversion VALU.
// ---------------------------------------------------------------------------
__global__ __launch_bounds__(256) void k1_gemm(const bf16_t* __restrict__ Ahi_g,
                                               const bf16_t* __restrict__ Alo_g,
                                               const bf16_t* __restrict__ EhiT,
                                               const bf16_t* __restrict__ EloT,
                                               float* __restrict__ L32,
                                               bf16_t* __restrict__ L16) {
  const int P = 40;  // tile pitch in halfs: 80B rows -> 16B aligned, 16 banks/16 rows
  __shared__ __align__(16) bf16_t sAh[2][128 * 40];
  __shared__ __align__(16) bf16_t sAl[2][128 * 40];
  __shared__ __align__(16) bf16_t sBh[2][128 * 40];  // [n][k] (pre-transposed src)
  __shared__ __align__(16) bf16_t sBl[2][128 * 40];

  const int t = threadIdx.x, lane = t & 31, w = t >> 5;
  const int n0 = blockIdx.x * 128, m0 = blockIdx.y * 128;
  const int wm = w >> 2, wn = w & 3;  // 2x4 wave grid

  v8f acc[4][2];
#pragma unroll
  for (int i = 0; i < 4; i++)
#pragma unroll
    for (int j = 0; j < 2; j++) { v8f z = V8F_ZERO; acc[i][j] = z; }

  // Stage one 32-deep K-slice of all four operand planes: 256 threads x
  // 4 async b128 copies each (128 rows x two 8-half chunks per plane, at halfs 0 and 32).
  auto stage = [&](int p, int k0) {
    const int r = t >> 1, c = (t & 1) * 32;
    async_copy_b128(lds_off(&sAh[p][r * P + c]),
                    Ahi_g + (size_t)(m0 + r) * NN + k0 + c);
    async_copy_b128(lds_off(&sAl[p][r * P + c]),
                    Alo_g + (size_t)(m0 + r) * NN + k0 + c);
    async_copy_b128(lds_off(&sBh[p][r * P + c]),
                    EhiT + (size_t)(n0 + r) * NN + k0 + c);
    async_copy_b128(lds_off(&sBl[p][r * P + c]),
                    EloT + (size_t)(n0 + r) * NN + k0 + c);
  };

  stage(0, 0);
  for (int k0 = 0; k0 < NN; k0 += 32) {
    const int p = (k0 >> 5) & 1;
    wait_async0();     // this wave's copies for buffer p have landed
    __syncthreads();   // ... and everyone else's
    if (k0 + 32 < NN) stage(p ^ 1, k0 + 32);  // prefetch next slice, overlapped

#pragma unroll
    for (int mt = 0; mt < 4; mt++) {
      const int ar = (wm * 64 + mt * 16 + (lane & 15)) * P;
      v16bf ah = load_afrag(&sAh[p][ar], lane);
      v16bf al = load_afrag(&sAl[p][ar], lane);
#pragma unroll
      for (int nt = 0; nt < 2; nt++) {
        const int br = (wn * 32 + nt * 16 + (lane & 15)) * P;
        v16bf bh = load_bfrag(&sBh[p][br], lane);
        v16bf bl = load_bfrag(&sBl[p][br], lane);
        acc[mt][nt] = wmma_bf16(ah, bh, acc[mt][nt]);  // hi*hi
        acc[mt][nt] = wmma_bf16(ah, bl, acc[mt][nt]);  // hi*lo
        acc[mt][nt] = wmma_bf16(al, bh, acc[mt][nt]);  // lo*hi
      }
    }
    // Buffer p is only overwritten by the stage() issued after the *next*
    // iteration's barrier, i.e. after every wave consumed its fragments.
  }

  // Epilogue: C/D layout -> lane holds col n, rows m = r + 8*lane[4]
#pragma unroll
  for (int mt = 0; mt < 4; mt++)
#pragma unroll
    for (int nt = 0; nt < 2; nt++) {
      int nn = n0 + wn * 32 + nt * 16 + (lane & 15);
      int mbase = m0 + wm * 64 + mt * 16 + ((lane & 16) ? 8 : 0);
#pragma unroll
      for (int r = 0; r < 8; r++) {
        float vv = acc[mt][nt][r];
        size_t o = (size_t)(mbase + r) * HH + nn;
        L32[o] = vv;
        L16[o] = (bf16_t)vv;
      }
    }
}

// ---------------------------------------------------------------------------
// k2: one block per (b,s) pair: gather -> score WMMA -> dual softmax ->
//     eq/ea WMMA -> tq/ta = (x - e)^2 written as f32 + bf16
// ---------------------------------------------------------------------------
__global__ __launch_bounds__(256) void k2_align(const float* __restrict__ L32,
                                                const bf16_t* __restrict__ L16,
                                                const int* __restrict__ q_idx,
                                                const int* __restrict__ a_idx,
                                                bf16_t* __restrict__ QTg,
                                                bf16_t* __restrict__ ATg,
                                                bf16_t* __restrict__ tq_bf,
                                                bf16_t* __restrict__ ta_bf,
                                                float* __restrict__ tq_f,
                                                float* __restrict__ ta_f) {
  __shared__ __align__(16) float  score[WW * WW];       // [w][v]
  __shared__ __align__(16) bf16_t WQ[WW * 72];          // wq row-major [w][v]
  __shared__ __align__(16) bf16_t WAT[WW * 72];         // wa^T row-major [v][w]
  __shared__ int qrow[WW], arow[WW];

  const int t = threadIdx.x, lane = t & 31, w = t >> 5;
  const int pair = blockIdx.x;  // b*S + s

  if (t < WW) {
    qrow[t] = q_idx[pair * WW + t];
    arow[t] = a_idx[pair * WW + t];
  }
  __syncthreads();

  bf16_t* QT = QTg + (size_t)pair * HH * WW;  // [h][w] bf16
  bf16_t* AT = ATg + (size_t)pair * HH * WW;

  // Phase 1: transposed gathers into global scratch (L2-resident)
  for (int i = t; i < WW * HH; i += 256) {
    int ww = i >> 9, h = i & 511;
    QT[h * WW + ww] = (bf16_t)L32[(size_t)qrow[ww] * HH + h];
    AT[h * WW + ww] = (bf16_t)L32[(size_t)arow[ww] * HH + h];
  }
  __threadfence();

  // Phase 2: score[w][v] = Q . Aenc^T, fragments streamed from L2-resident L16
  {
    const int mt = w & 3, ntb = (w >> 2) * 2;
    const bf16_t* qp = L16 + (size_t)qrow[mt * 16 + (lane & 15)] * HH;
    const bf16_t* b0 = L16 + (size_t)arow[(ntb + 0) * 16 + (lane & 15)] * HH;
    const bf16_t* b1 = L16 + (size_t)arow[(ntb + 1) * 16 + (lane & 15)] * HH;
    v8f s0 = V8F_ZERO, s1 = V8F_ZERO;
    for (int k0 = 0; k0 < HH; k0 += 32) {
      v16bf a = load_afrag(qp + k0, lane);
      s0 = wmma_bf16(a, load_bfrag(b0 + k0, lane), s0);
      s1 = wmma_bf16(a, load_bfrag(b1 + k0, lane), s1);
    }
#pragma unroll
    for (int r = 0; r < 8; r++) {
      int m = mt * 16 + r + ((lane & 16) ? 8 : 0);
      score[m * WW + (ntb + 0) * 16 + (lane & 15)] = s0[r];
      score[m * WW + (ntb + 1) * 16 + (lane & 15)] = s1[r];
    }
  }
  __syncthreads();

  // Phase 3: softmax over v (rows -> WQ) and over w (cols -> WAT, stored transposed)
  if (t < WW) {
    float mx = -1e30f;
    for (int v = 0; v < WW; v++) mx = fmaxf(mx, score[t * WW + v]);
    float sm = 0.f;
    for (int v = 0; v < WW; v++) sm += __expf(score[t * WW + v] - mx);
    float inv = 1.f / sm;
    for (int v = 0; v < WW; v++)
      WQ[t * 72 + v] = (bf16_t)(__expf(score[t * WW + v] - mx) * inv);
  } else if (t < 2 * WW) {
    int c = t - WW;
    float mx = -1e30f;
    for (int u = 0; u < WW; u++) mx = fmaxf(mx, score[u * WW + c]);
    float sm = 0.f;
    for (int u = 0; u < WW; u++) sm += __expf(score[u * WW + c] - mx);
    float inv = 1.f / sm;
    for (int u = 0; u < WW; u++)
      WAT[c * 72 + u] = (bf16_t)(__expf(score[u * WW + c] - mx) * inv);
  }
  __syncthreads();

  // Phase 4: eq = wq @ Aenc (B-frag from AT), ea = wa^T @ Q (B-frag from QT);
  //          fused residual square.
  for (int sel = 0; sel < 2; sel++) {
    const bf16_t* Af = sel ? WAT : WQ;      // LDS, row-major over K=64
    const bf16_t* Bt = sel ? QT : AT;       // global [h][w/v]
    const int*    rs = sel ? arow : qrow;   // residual source rows
    float*  outf = (sel ? ta_f : tq_f) + (size_t)pair * WW * HH;
    bf16_t* outb = (sel ? ta_bf : tq_bf) + (size_t)pair * WW * HH;
    for (int tile = w; tile < 128; tile += 8) {
      int mt = tile >> 5, nt = tile & 31;
      const bf16_t* ar = Af + (mt * 16 + (lane & 15)) * 72;
      const bf16_t* br = Bt + (nt * 16 + (lane & 15)) * WW;
      v8f acc = V8F_ZERO;
      acc = wmma_bf16(load_afrag(ar, lane),      load_bfrag(br, lane),      acc);
      acc = wmma_bf16(load_afrag(ar + 32, lane), load_bfrag(br + 32, lane), acc);
#pragma unroll
      for (int r = 0; r < 8; r++) {
        int m = mt * 16 + r + ((lane & 16) ? 8 : 0);
        int n = nt * 16 + (lane & 15);
        float xv = L32[(size_t)rs[m] * HH + n];
        float d = xv - acc[r];
        float tv = d * d;
        outf[m * HH + n] = tv;
        outb[m * HH + n] = (bf16_t)tv;
      }
    }
  }
}

// ---------------------------------------------------------------------------
// k3: one block per (b,s,which): highway MLP via dual-accumulator WMMA,
//     fused relu/sigmoid/residual and sum over W -> interleaved feat
// ---------------------------------------------------------------------------
__global__ __launch_bounds__(256) void k3_highway(const bf16_t* __restrict__ tq_bf,
                                                  const bf16_t* __restrict__ ta_bf,
                                                  const float* __restrict__ tq_f,
                                                  const float* __restrict__ ta_f,
                                                  const bf16_t* __restrict__ WhT,
                                                  const bf16_t* __restrict__ WtT,
                                                  const float* __restrict__ bh,
                                                  const float* __restrict__ bt,
                                                  float* __restrict__ feat) {
  const int t = threadIdx.x, lane = t & 31, w = t >> 5;
  const int blk = blockIdx.x;
  const int which = blk & 1;
  const int s = (blk >> 1) % SS;
  const int b = blk / (2 * SS);
  const int pair = b * SS + s;

  const bf16_t* Xb = (which ? ta_bf : tq_bf) + (size_t)pair * WW * HH;
  const float*  Xf = (which ? ta_f  : tq_f ) + (size_t)pair * WW * HH;
  const bf16_t* Bh = WhT + (size_t)s * HH * HH;
  const bf16_t* Bt = WtT + (size_t)s * HH * HH;
  const float* bhv = bh + s * HH;
  const float* btv = bt + s * HH;

#pragma unroll
  for (int j = 0; j < 4; j++) {
    const int nt = w * 4 + j;
    const int g = nt * 16 + (lane & 15);
    const float bhg = bhv[g], btg = btv[g];
    const bf16_t* bhrow = Bh + (size_t)g * HH;
    const bf16_t* btrow = Bt + (size_t)g * HH;
    float sum = 0.f;
    for (int mt = 0; mt < 4; mt++) {
      const bf16_t* ar = Xb + (size_t)(mt * 16 + (lane & 15)) * HH;
      v8f hacc = V8F_ZERO, tacc = V8F_ZERO;
      for (int k0 = 0; k0 < HH; k0 += 32) {
        v16bf a = load_afrag(ar + k0, lane);
        hacc = wmma_bf16(a, load_bfrag(bhrow + k0, lane), hacc);
        tacc = wmma_bf16(a, load_bfrag(btrow + k0, lane), tacc);
      }
#pragma unroll
      for (int r = 0; r < 8; r++) {
        int m = mt * 16 + r + ((lane & 16) ? 8 : 0);
        float hv = fmaxf(hacc[r] + bhg, 0.f);
        float tv = 1.f / (1.f + __expf(-(tacc[r] + btg)));
        float xv = Xf[m * HH + g];
        sum += tv * hv + (1.f - tv) * xv;
      }
    }
    sum += __shfl_xor(sum, 16, 32);  // combine the two m-halves (same column g)
    if (lane < 16)
      feat[(size_t)b * (2 * SS * HH) + (size_t)(s * 2 + which) * HH + g] =
          sum * 0.57735026918962576f;  // 1/sqrt(S)
  }
}

// ---------------------------------------------------------------------------
// k4: logits = feat @ W_lin + b_lin ; log_softmax over 2 classes
// ---------------------------------------------------------------------------
__global__ __launch_bounds__(256) void k4_logits(const float* __restrict__ feat,
                                                 const float* __restrict__ Wlin,
                                                 const float* __restrict__ blin,
                                                 float* __restrict__ out) {
  __shared__ float r0[256], r1[256];
  const int t = threadIdx.x, b = blockIdx.x;
  float a0 = 0.f, a1 = 0.f;
  for (int i = t; i < 2 * SS * HH; i += 256) {
    float f = feat[(size_t)b * (2 * SS * HH) + i];
    a0 += f * Wlin[i * 2 + 0];
    a1 += f * Wlin[i * 2 + 1];
  }
  r0[t] = a0; r1[t] = a1;
  __syncthreads();
  for (int st = 128; st > 0; st >>= 1) {
    if (t < st) { r0[t] += r0[t + st]; r1[t] += r1[t + st]; }
    __syncthreads();
  }
  if (t == 0) {
    float l0 = r0[0] + blin[0], l1 = r1[0] + blin[1];
    float mx = fmaxf(l0, l1);
    float lse = mx + __logf(__expf(l0 - mx) + __expf(l1 - mx));
    out[b * 2 + 0] = l0 - lse;
    out[b * 2 + 1] = l1 - lse;
  }
}

// ---------------------------------------------------------------------------
extern "C" void kernel_launch(void* const* d_in, const int* in_sizes, int n_in,
                              void* d_out, int out_size, void* d_ws, size_t ws_size,
                              hipStream_t stream) {
  (void)in_sizes; (void)n_in; (void)out_size; (void)ws_size;
  const float* A    = (const float*)d_in[0];
  const float* emb  = (const float*)d_in[1];
  const float* Wh   = (const float*)d_in[2];
  const float* bh   = (const float*)d_in[3];
  const float* Wt   = (const float*)d_in[4];
  const float* bt   = (const float*)d_in[5];
  const float* Wlin = (const float*)d_in[6];
  const float* blin = (const float*)d_in[7];
  const int* q_idx  = (const int*)d_in[8];
  const int* a_idx  = (const int*)d_in[9];
  float* out = (float*)d_out;

  // Workspace carve-up (256B aligned)
  char* ws = (char*)d_ws;
  auto carve = [&](size_t bytes) {
    char* p = ws;
    ws += (bytes + 255) & ~(size_t)255;
    return p;
  };
  bf16_t* Ahi_g = (bf16_t*)carve((size_t)NN * NN * 2);            // 128 MB
  bf16_t* Alo_g = (bf16_t*)carve((size_t)NN * NN * 2);            // 128 MB
  bf16_t* EhiT  = (bf16_t*)carve((size_t)HH * NN * 2);            //   8 MB
  bf16_t* EloT  = (bf16_t*)carve((size_t)HH * NN * 2);            //   8 MB
  float*  L32   = (float*) carve((size_t)NN * HH * 4);            //  16 MB
  bf16_t* L16   = (bf16_t*)carve((size_t)NN * HH * 2);            //   8 MB
  bf16_t* WhT   = (bf16_t*)carve((size_t)SS * HH * HH * 2);       // 1.5 MB
  bf16_t* WtT   = (bf16_t*)carve((size_t)SS * HH * HH * 2);       // 1.5 MB
  bf16_t* QTg   = (bf16_t*)carve((size_t)BB * SS * HH * WW * 2);  // 12.6 MB
  bf16_t* ATg   = (bf16_t*)carve((size_t)BB * SS * HH * WW * 2);
  bf16_t* tq_bf = (bf16_t*)carve((size_t)BB * SS * WW * HH * 2);
  bf16_t* ta_bf = (bf16_t*)carve((size_t)BB * SS * WW * HH * 2);
  float*  tq_f  = (float*) carve((size_t)BB * SS * WW * HH * 4);  // 25 MB
  float*  ta_f  = (float*) carve((size_t)BB * SS * WW * HH * 4);
  float*  feat  = (float*) carve((size_t)BB * 2 * SS * HH * 4);

  hipLaunchKernelGGL(kA_split, dim3((size_t)NN * NN / 4 / 256), dim3(256), 0, stream,
                     A, Ahi_g, Alo_g);
  hipLaunchKernelGGL(kE_splitT, dim3((size_t)HH * NN / 4 / 256), dim3(256), 0, stream,
                     emb, EhiT, EloT);
  hipLaunchKernelGGL(k0_transposeW, dim3(2 * SS * HH * HH / 256), dim3(256), 0, stream,
                     Wh, Wt, WhT, WtT);
  hipLaunchKernelGGL(k1_gemm, dim3(HH / 128, NN / 128), dim3(256), 0, stream,
                     Ahi_g, Alo_g, EhiT, EloT, L32, L16);
  hipLaunchKernelGGL(k2_align, dim3(BB * SS), dim3(256), 0, stream,
                     L32, L16, q_idx, a_idx, QTg, ATg, tq_bf, ta_bf, tq_f, ta_f);
  hipLaunchKernelGGL(k3_highway, dim3(BB * SS * 2), dim3(256), 0, stream,
                     tq_bf, ta_bf, tq_f, ta_f, WhT, WtT, bh, bt, feat);
  hipLaunchKernelGGL(k4_logits, dim3(BB), dim3(256), 0, stream,
                     feat, Wlin, blin, out);
}
